// TCN_8461085573315
// MI455X (gfx1250) — matrix operand order repaired and checked by reading, not compiled, not run on hardware
//
#include <hip/hip_runtime.h>
#include <hip/hip_bf16.h>

// ---------------------------------------------------------------------------
// Problem constants (from reference): S=1024, B=16, Cin=Cout=256, K=3
// ---------------------------------------------------------------------------
#define SS   1024
#define BB   16
#define CIN  256
#define COUT 256
#define KW   3
#define KDIM (CIN * KW)            // 768, GEMM K with index k*CIN+cin
#define NSB  (SS * BB)             // 16384
#define ATT  ((SS - 1) * SS * BB)  // 1023*1024*16 per attention tensor
#define NPACK (16 * 24 * 32 * 16)  // packed bf16 weights: kk*ct*lane*j = 196608

typedef __bf16 bf16_t;
typedef bf16_t v16bf __attribute__((ext_vector_type(16)));
typedef float  v8f   __attribute__((ext_vector_type(8)));
typedef int    v4i   __attribute__((ext_vector_type(4)));

#define AS1 __attribute__((address_space(1)))   // global
#define AS3 __attribute__((address_space(3)))   // LDS

__device__ __forceinline__ float ht_f(float v) {
    // 0.5 * (clip(v*10, -1, 1) + 1)
    return 0.5f * (fminf(fmaxf(v * 10.0f, -1.0f), 1.0f) + 1.0f);
}

// ---- CDNA5 async global->LDS copy (ASYNCcnt-tracked) ----------------------
__device__ __forceinline__ void async_g2l_b128(const void* gsrc, void* ldst) {
    __builtin_amdgcn_global_load_async_to_lds_b128(
        (AS1 v4i*)(void*)gsrc, (AS3 v4i*)ldst, 0, 0);
}

template <int N>
__device__ __forceinline__ void wait_async_le() {
#if __has_builtin(__builtin_amdgcn_s_wait_asynccnt)
    __builtin_amdgcn_s_wait_asynccnt(N);
#else
    asm volatile("s_wait_asynccnt %0" :: "i"(N) : "memory");
#endif
}

__device__ __forceinline__ void wait_ds0() {
    asm volatile("s_wait_dscnt 0x0" ::: "memory");
}

// ---------------------------------------------------------------------------
// Kernel P: pre-pack conv weights into the WMMA B-matrix lane layout (bf16).
// B tile for (ct, kk): 32x16; lane L holds column n=L&15, K-locals
// half*8 + (j&7) + 16*(j>>3) (j=0..15), global K = kk*32 + local, k = K/256, cin = K%256.
// Packed index: ((kk*16 + ct)*32 + lane)*16 + j  (kk-major so one stage is
// a single contiguous 16 KB block in memory).
// ---------------------------------------------------------------------------
__global__ __launch_bounds__(256)
void pack_w(const float* __restrict__ w, bf16_t* __restrict__ wp)
{
    const int t = blockIdx.x * 256 + threadIdx.x;   // < NPACK
    const int j    = t & 15;
    const int lane = (t >> 4) & 31;
    const int rest = t >> 9;                        // kk*16 + ct
    const int ct   = rest & 15;
    const int kk   = rest >> 4;
    const int half = lane >> 4;
    const int nidx = lane & 15;
    const int cout = ct * 16 + nidx;
    const int k    = kk >> 3;
    const int cin  = (kk & 7) * 32 + (half << 3) + (j & 7) + ((j >> 3) << 4);
    wp[t] = (bf16_t)w[cout * KDIM + cin * KW + k];
}

// ---------------------------------------------------------------------------
// Kernel A: causal conv1d + bias + relu + linear -> x_output (S*B floats)
// Implicit GEMM, one wave per 16 sb rows. kk-outer / ct-inner.
// B operand is double-buffered in LDS via async global->LDS copies so the
// 16 WMMAs of chunk kk overlap the fetch of chunk kk+1 (ASYNCcnt pipeline).
// ---------------------------------------------------------------------------
__global__ __launch_bounds__(32)
void conv_lin_wmma(const float* __restrict__ x,
                   const bf16_t* __restrict__ wpack,
                   const float* __restrict__ cb,
                   const float* __restrict__ lw,
                   const float* __restrict__ lb,
                   float* __restrict__ xo)
{
    // 2 x 16 KB double buffer: [parity][ct][lane][16 bf16]
    __shared__ __align__(32) bf16_t lbuf[2][16][32][16];

    const int lane = threadIdx.x & 31;
    const int nidx = lane & 15;     // A row within tile / B column within tile
    const int half = lane >> 4;     // lane group 0/1 (selects K sub-ranges)
    const int tile = blockIdx.x;    // 1024 tiles of 16 sb rows
    const int sb   = tile * 16 + nidx;

    // stage all 16 B tiles of K-chunk kk into LDS buffer p (32 async b128)
    auto stage = [&](int kk, int p) {
        wait_ds0();   // LDS reads of buffer p must be done before overwrite
        const bf16_t* src = wpack + ((long)kk * 16 * 32) * 16; // 16 KB block
#pragma unroll
        for (int ct = 0; ct < 16; ++ct) {
            const bf16_t* g = src + (ct * 32 + lane) * 16;
            bf16_t*       l = &lbuf[p][ct][lane][0];
            async_g2l_b128(g, l);          // bytes 0..15
            async_g2l_b128(g + 8, l + 8);  // bytes 16..31
        }
    };

    v8f cacc[16];
#pragma unroll
    for (int ct = 0; ct < 16; ++ct) cacc[ct] = (v8f){};

    stage(0, 0);

    for (int kk = 0; kk < KDIM / 32; ++kk) {   // 24 K-chunks of 32
        const int p = kk & 1;
        if (kk < KDIM / 32 - 1) {
            stage(kk + 1, 1 - p);
            // wait only for chunk kk's copies; next stage's 32 stay in flight
            wait_async_le<32>();
        } else {
            wait_async_le<0>();
        }

        const int  k     = kk >> 3;            // conv tap 0..2
        const int  cgrp  = (kk & 7) * 32;      // cin base of this chunk
        const int  row   = sb + (k - 2) * BB;  // shifted (s,b) row
        const bool valid = row >= 0;           // causal left padding
        const int  crow  = valid ? row : 0;    // clamp -> loads always legal

        // ---- A: 16x32 bf16 tile of x patches (4x global_load_b128) -------
        // lanes 0-15 hold K = {0..7,16..23}, lanes 16-31 K = {8..15,24..31}
        const float* ap = x + (long)crow * CIN + cgrp + (half << 3);
        const float4 f0 = *(const float4*)(ap);
        const float4 f1 = *(const float4*)(ap + 4);
        const float4 f2 = *(const float4*)(ap + 16);
        const float4 f3 = *(const float4*)(ap + 20);

        v16bf a;
        a[0]  = (bf16_t)(valid ? f0.x : 0.0f);
        a[1]  = (bf16_t)(valid ? f0.y : 0.0f);
        a[2]  = (bf16_t)(valid ? f0.z : 0.0f);
        a[3]  = (bf16_t)(valid ? f0.w : 0.0f);
        a[4]  = (bf16_t)(valid ? f1.x : 0.0f);
        a[5]  = (bf16_t)(valid ? f1.y : 0.0f);
        a[6]  = (bf16_t)(valid ? f1.z : 0.0f);
        a[7]  = (bf16_t)(valid ? f1.w : 0.0f);
        a[8]  = (bf16_t)(valid ? f2.x : 0.0f);
        a[9]  = (bf16_t)(valid ? f2.y : 0.0f);
        a[10] = (bf16_t)(valid ? f2.z : 0.0f);
        a[11] = (bf16_t)(valid ? f2.w : 0.0f);
        a[12] = (bf16_t)(valid ? f3.x : 0.0f);
        a[13] = (bf16_t)(valid ? f3.y : 0.0f);
        a[14] = (bf16_t)(valid ? f3.z : 0.0f);
        a[15] = (bf16_t)(valid ? f3.w : 0.0f);

        // ---- 16 WMMAs, B tiles read from LDS (ds_load_b128 pairs) ---------
#pragma unroll
        for (int ct = 0; ct < 16; ++ct) {
            const v16bf bm = *(const v16bf*)(&lbuf[p][ct][lane][0]);
            cacc[ct] = __builtin_amdgcn_wmma_f32_16x16x32_bf16(
                           false, a, false, bm, (short)0, cacc[ct], false, false);
        }
    }

    // bias + relu + lin_w reduction over cout (each lane owns cout = ct*16+nidx)
    float acc[8];
#pragma unroll
    for (int r = 0; r < 8; ++r) acc[r] = 0.0f;
#pragma unroll
    for (int ct = 0; ct < 16; ++ct) {
        const float cbv = cb[ct * 16 + nidx];
        const float lwv = lw[ct * 16 + nidx];
#pragma unroll
        for (int r = 0; r < 8; ++r) {
            float v = cacc[ct][r] + cbv;
            v = v > 0.0f ? v : 0.0f;
            acc[r] += v * lwv;
        }
    }

    // reduce the 16 lanes holding the 16 couts of each row group
#pragma unroll
    for (int r = 0; r < 8; ++r) {
#pragma unroll
        for (int m = 1; m < 16; m <<= 1)
            acc[r] += __shfl_xor(acc[r], m, 16);
    }
    if (nidx == 0) {
        const float lb0 = lb[0];
#pragma unroll
        for (int r = 0; r < 8; ++r)
            xo[tile * 16 + (half << 3) + r] = acc[r] + lb0;   // M = r + 8*half
    }
}

// ---------------------------------------------------------------------------
// Kernel B: scan tables. Single block (16K elements).
//   xs[j,b]   = x_output[j-1,b] (0 at j=0)            (shifter_down @ x_output)
//   h[j,b]    = ht(xs - xo)                           (square_2 term, fwd)
//   xatt[j,b] = xo - min(xo) + 0.05
//   xsr[m,b]  = xs[S-1-m,b] ; hc[m,b] = ht(xsr - xo[m])   (context branch)
// ---------------------------------------------------------------------------
__global__ __launch_bounds__(1024)
void prep(const float* __restrict__ xo,
          float* __restrict__ xs, float* __restrict__ h,
          float* __restrict__ xatt,
          float* __restrict__ xsr, float* __restrict__ hc)
{
    __shared__ float red[1024];
    const int tid = threadIdx.x;

    float lmin = 3.4e38f;
    for (int idx = tid; idx < NSB; idx += 1024)
        lmin = fminf(lmin, xo[idx]);
    red[tid] = lmin;
    __syncthreads();
    for (int off = 512; off > 0; off >>= 1) {
        if (tid < off) red[tid] = fminf(red[tid], red[tid + off]);
        __syncthreads();
    }
    const float gmin = red[0];

    for (int idx = tid; idx < NSB; idx += 1024) {
        const float xov = xo[idx];
        const float xsv = (idx >= BB) ? xo[idx - BB] : 0.0f;  // j-1 row
        xs[idx]   = xsv;
        h[idx]    = ht_f(xsv - xov);
        xatt[idx] = xov - gmin + 0.05f;
    }
    __syncthreads();
    for (int idx = tid; idx < NSB; idx += 1024) {
        const int m = idx >> 4, b = idx & 15;
        const float v = xs[(SS - 1 - m) * BB + b];
        xsr[idx] = v;
        hc[idx]  = ht_f(v - xo[idx]);
    }
}

// ---------------------------------------------------------------------------
// Kernel C: forward attention.  thread = i*16+b  (i in 0..1022)
// span_p[i,j] = 0 (j<=i), 1 (j==i+1), prod_{j'=i+2..j}(1 - ht(xs[j']-xo[i])*h[j'])
// attn = span*xatt[j] / sum_j
// ---------------------------------------------------------------------------
__global__ __launch_bounds__(256)
void attn_p(const float* __restrict__ xo,
            const float* __restrict__ xs,
            const float* __restrict__ h,
            const float* __restrict__ xatt,
            float* __restrict__ out)
{
    const int t = blockIdx.x * blockDim.x + threadIdx.x;
    if (t >= (SS - 1) * BB) return;
    const int i = t >> 4, b = t & 15;
    const float xoi = xo[i * BB + b];

    // pass 1: normalization sum
    float sum = xatt[(i + 1) * BB + b];   // span == 1 at j = i+1
    float p = 1.0f;
    for (int j = i + 2; j < SS; ++j) {
        const int idx = j * BB + b;
        if (j + 64 < SS) __builtin_prefetch(&xs[idx + 64 * BB], 0, 1);
        p *= 1.0f - ht_f(xs[idx] - xoi) * h[idx];
        sum += p * xatt[idx];
    }
    const float inv = 1.0f / sum;

    // pass 2: write all j (d_out is poisoned -> zeros must be written too)
    float* row = out + (long)i * SS * BB + b;
    for (int j = 0; j <= i; ++j) row[j * BB] = 0.0f;
    row[(i + 1) * BB] = xatt[(i + 1) * BB + b] * inv;
    p = 1.0f;
    for (int j = i + 2; j < SS; ++j) {
        const int idx = j * BB + b;
        p *= 1.0f - ht_f(xs[idx] - xoi) * h[idx];
        row[j * BB] = p * xatt[idx] * inv;
    }
}

// ---------------------------------------------------------------------------
// Kernel D: context attention (reversed branch, un-reversed indices).
// span_c[i,j] = 0 (j>i), 1 (j in {i,i-1}),
//               prod_{m=j..i-2}(1 - ht(xsr[m]-xo[i])*hc[m])  (j<=i-2)
// ---------------------------------------------------------------------------
__global__ __launch_bounds__(256)
void attn_c(const float* __restrict__ xo,
            const float* __restrict__ xsr,
            const float* __restrict__ hc,
            const float* __restrict__ xatt,
            float* __restrict__ out)
{
    const int t = blockIdx.x * blockDim.x + threadIdx.x;
    if (t >= (SS - 1) * BB) return;
    const int i = t >> 4, b = t & 15;
    const float xoi = xo[i * BB + b];

    float sum = xatt[i * BB + b];
    if (i >= 1) sum += xatt[(i - 1) * BB + b];
    float p = 1.0f;
    for (int j = i - 2; j >= 0; --j) {
        const int idx = j * BB + b;
        if (j >= 64) __builtin_prefetch(&xsr[idx - 64 * BB], 0, 1);
        p *= 1.0f - ht_f(xsr[idx] - xoi) * hc[idx];
        sum += p * xatt[idx];
    }
    const float inv = 1.0f / sum;

    float* row = out + (long)i * SS * BB + b;
    for (int j = i + 1; j < SS; ++j) row[j * BB] = 0.0f;
    row[i * BB] = xatt[i * BB + b] * inv;
    if (i >= 1) row[(i - 1) * BB] = xatt[(i - 1) * BB + b] * inv;
    p = 1.0f;
    for (int j = i - 2; j >= 0; --j) {
        const int idx = j * BB + b;
        p *= 1.0f - ht_f(xsr[idx] - xoi) * hc[idx];
        row[j * BB] = p * xatt[idx] * inv;
    }
}

// ---------------------------------------------------------------------------
// Kernel E: pass-through outputs (seq_len_data bits, reg_len = 0)
// ---------------------------------------------------------------------------
__global__ __launch_bounds__(32)
void tail(const int* __restrict__ seq, float* __restrict__ out)
{
    const int t = threadIdx.x;
    if (t < BB)       out[t]  = __int_as_float(seq[t]);
    else if (t == BB) out[BB] = 0.0f;   // reg_len
}

// ---------------------------------------------------------------------------
extern "C" void kernel_launch(void* const* d_in, const int* in_sizes, int n_in,
                              void* d_out, int out_size, void* d_ws, size_t ws_size,
                              hipStream_t stream)
{
    const float* x   = (const float*)d_in[0];
    const int*   seq = (const int*)  d_in[1];
    const float* cw  = (const float*)d_in[2];
    const float* cb  = (const float*)d_in[3];
    const float* lw  = (const float*)d_in[4];
    const float* lb  = (const float*)d_in[5];
    float* out = (float*)d_out;
    float* ws  = (float*)d_ws;

    float*  xo    = ws;             // S*B
    float*  xs    = ws + 1 * NSB;   // shifted-down x_output
    float*  h     = ws + 2 * NSB;   // ht(xs - xo)
    float*  xatt  = ws + 3 * NSB;   // xo - min + 0.05
    float*  xsr   = ws + 4 * NSB;   // xs reflected
    float*  hc    = ws + 5 * NSB;   // ht(xsr - xo)
    bf16_t* wpack = (bf16_t*)(ws + 6 * NSB);   // 384 KB packed weights (32B aligned)

    pack_w<<<NPACK / 256, 256, 0, stream>>>(cw, wpack);
    conv_lin_wmma<<<NSB / 16, 32, 0, stream>>>(x, wpack, cb, lw, lb, xo);
    prep<<<1, 1024, 0, stream>>>(xo, xs, h, xatt, xsr, hc);

    const int nthreads = (SS - 1) * BB;           // 16368
    const int blocks   = (nthreads + 255) / 256;
    attn_p<<<blocks, 256, 0, stream>>>(xo, xs, h, xatt, out);
    attn_c<<<blocks, 256, 0, stream>>>(xo, xsr, hc, xatt, out + (long)ATT);
    tail<<<1, 32, 0, stream>>>(seq, out + 2L * ATT);
}
